// KuramotoSystemV2_91001767068244
// MI455X (gfx1250) — matrix operand order repaired and checked by reading, not compiled, not run on hardware
//
#include <hip/hip_runtime.h>
#include <math.h>

typedef __attribute__((ext_vector_type(16))) _Float16 v16h;
typedef __attribute__((ext_vector_type(8)))  _Float16 v8h;
typedef __attribute__((ext_vector_type(8)))  float    v8f;
typedef __attribute__((ext_vector_type(4)))  float    v4f;

#define WAVE 32
#define TWO_PI_F 6.2831855f   // np.float32(2*pi)

// ---------------------------------------------------------------------------
// Kernel 1: row-normalize embeddings (f32 -> f16).  One wave32 per row of D.
// ---------------------------------------------------------------------------
__global__ void normalize_rows_f16(const float* __restrict__ E,
                                   _Float16* __restrict__ Eh,
                                   int rows, int D) {
  const int wid  = threadIdx.x >> 5;
  const int lane = threadIdx.x & 31;
  const int row  = blockIdx.x * (blockDim.x >> 5) + wid;
  if (row >= rows) return;

  const float* src = E + (size_t)row * D;
  float ss = 0.0f;
  for (int d = lane * 8; d < D; d += WAVE * 8) {
    v4f a = *(const v4f*)(src + d);
    v4f b = *(const v4f*)(src + d + 4);
#pragma unroll
    for (int t = 0; t < 4; ++t) ss = fmaf(a[t], a[t], fmaf(b[t], b[t], ss));
  }
#pragma unroll
  for (int off = 16; off; off >>= 1) ss += __shfl_xor(ss, off, 32);

  const float scale = 1.0f / fmaxf(sqrtf(ss), 1e-12f);

  _Float16* dst = Eh + (size_t)row * D;
  for (int d = lane * 8; d < D; d += WAVE * 8) {
    v4f a = *(const v4f*)(src + d);
    v4f b = *(const v4f*)(src + d + 4);
    v8h o;
#pragma unroll
    for (int t = 0; t < 4; ++t) {
      o[t]     = (_Float16)(a[t] * scale);
      o[4 + t] = (_Float16)(b[t] * scale);
    }
    *(v8h*)(dst + d) = o;
  }
}

// ---------------------------------------------------------------------------
// Kernel 2: C[b] = relu(Eh[b] @ Eh[b]^T)  via v_wmma_f32_16x16x32_f16.
// One wave computes a 16x64 output strip (A fragment reused across 4 B frags).
// Since C = E*E^T, both A and B fragments are contiguous loads from Eh.
// ---------------------------------------------------------------------------
__global__ void coupling_wmma(const _Float16* __restrict__ Eh,
                              _Float16* __restrict__ Ch,
                              int B, int Nn, int D) {
  const int wid  = threadIdx.x >> 5;
  const int lane = threadIdx.x & 31;
  const int gw   = blockIdx.x * (blockDim.x >> 5) + wid;

  const int rtiles  = Nn >> 4;          // 16-row tiles
  const int cblocks = Nn >> 6;          // 64-col strips
  const int perB    = rtiles * cblocks;

  const int b = gw / perB;
  if (b >= B) return;                   // wave-uniform: EXEC stays all-ones
  const int rem = gw - b * perB;
  const int rt  = rem / cblocks;
  const int cb  = rem - rt * cblocks;
  const int r0  = rt << 4;
  const int c0  = cb << 6;

  const int m = lane & 15;              // row within A tile / col within B tile
  const int h = lane >> 4;              // half-wave selector

  const _Float16* Arow  = Eh + ((size_t)b * Nn + (r0 + m)) * D;
  const _Float16* Bbase = Eh + (size_t)b * Nn * D;

  v8f acc0 = {}, acc1 = {}, acc2 = {}, acc3 = {};

  for (int k0 = 0; k0 < D; k0 += 32) {
    // A fragment (16x32 f16): lane(m,h) holds K [16h,16h+16)
    v16h a = *(const v16h*)(Arow + k0 + 16 * h);

    // B fragments (32x16 f16): lane(n=m,h) holds column n, K [16h,16h+16)
    v16h bf0 = *(const v16h*)(Bbase + (size_t)(c0 +  0 + m) * D + k0 + 16 * h);
    v16h bf1 = *(const v16h*)(Bbase + (size_t)(c0 + 16 + m) * D + k0 + 16 * h);
    v16h bf2 = *(const v16h*)(Bbase + (size_t)(c0 + 32 + m) * D + k0 + 16 * h);
    v16h bf3 = *(const v16h*)(Bbase + (size_t)(c0 + 48 + m) * D + k0 + 16 * h);

    acc0 = __builtin_amdgcn_wmma_f32_16x16x32_f16(false, a, false, bf0, (short)0, acc0, false, false);
    acc1 = __builtin_amdgcn_wmma_f32_16x16x32_f16(false, a, false, bf1, (short)0, acc1, false, false);
    acc2 = __builtin_amdgcn_wmma_f32_16x16x32_f16(false, a, false, bf2, (short)0, acc2, false, false);
    acc3 = __builtin_amdgcn_wmma_f32_16x16x32_f16(false, a, false, bf3, (short)0, acc3, false, false);
  }

  // D layout: lane(n=m,h), VGPR i -> element (row = i + 8h, col = n)
  _Float16* Cb = Ch + (size_t)b * Nn * Nn;
#pragma unroll
  for (int i = 0; i < 8; ++i) {
    const size_t base = (size_t)(r0 + i + 8 * h) * Nn + c0 + m;
    Cb[base +  0] = (_Float16)fmaxf(acc0[i], 0.0f);
    Cb[base + 16] = (_Float16)fmaxf(acc1[i], 0.0f);
    Cb[base + 32] = (_Float16)fmaxf(acc2[i], 0.0f);
    Cb[base + 48] = (_Float16)fmaxf(acc3[i], 0.0f);
  }
}

// ---------------------------------------------------------------------------
// Kernel 3a: cf = cos(theta), sf = sin(theta)  (all B*N phases)
// ---------------------------------------------------------------------------
__global__ void cossin_kernel(const float* __restrict__ theta,
                              float* __restrict__ cf, float* __restrict__ sf,
                              int n) {
  const int i = blockIdx.x * blockDim.x + threadIdx.x;
  if (i < n) {
    float s, c;
    sincosf(theta[i], &s, &c);
    cf[i] = c;
    sf[i] = s;
  }
}

// ---------------------------------------------------------------------------
// Kernel 3b: one wave32 per (b,i):
//   sum = sin(th_i)*(C_row . cos) - cos(th_i)*(C_row . sin)
//   th_i = fmod(th_i + dt*(omega_i + K/N * sum), 2*pi)
// C row (fp16, L2-resident) streamed with b128 loads; f32 accumulation.
// ---------------------------------------------------------------------------
__global__ void kuramoto_step(const _Float16* __restrict__ Ch,
                              const float* __restrict__ cf,
                              const float* __restrict__ sf,
                              const float* __restrict__ omega,
                              float* __restrict__ theta,
                              int B, int Nn, float k_over_n, float dt) {
  const int wid  = threadIdx.x >> 5;
  const int lane = threadIdx.x & 31;
  const int gw   = blockIdx.x * (blockDim.x >> 5) + wid;

  const int b = gw / Nn;
  if (b >= B) return;
  const int i = gw - b * Nn;

  const _Float16* Crow = Ch + ((size_t)b * Nn + i) * Nn;
  const float*    cb   = cf + (size_t)b * Nn;
  const float*    sb   = sf + (size_t)b * Nn;

  float dc = 0.0f, ds = 0.0f;
  for (int j0 = lane * 8; j0 < Nn; j0 += WAVE * 8) {
    v8h cv = *(const v8h*)(Crow + j0);
    v4f c0 = *(const v4f*)(cb + j0);
    v4f c1 = *(const v4f*)(cb + j0 + 4);
    v4f s0 = *(const v4f*)(sb + j0);
    v4f s1 = *(const v4f*)(sb + j0 + 4);
#pragma unroll
    for (int t = 0; t < 4; ++t) {
      const float w0 = (float)cv[t];
      const float w1 = (float)cv[4 + t];
      dc = fmaf(w0, c0[t], dc);
      ds = fmaf(w0, s0[t], ds);
      dc = fmaf(w1, c1[t], dc);
      ds = fmaf(w1, s1[t], ds);
    }
  }
#pragma unroll
  for (int off = 16; off; off >>= 1) {
    dc += __shfl_xor(dc, off, 32);
    ds += __shfl_xor(ds, off, 32);
  }

  if (lane == 0) {
    const float si  = sb[i];
    const float ci  = cb[i];
    const float sum = si * dc - ci * ds;
    const float dth = omega[i] + k_over_n * sum;
    const size_t idx = (size_t)b * Nn + i;
    theta[idx] = fmodf(theta[idx] + dt * dth, TWO_PI_F);
  }
}

// ---------------------------------------------------------------------------
extern "C" void kernel_launch(void* const* d_in, const int* in_sizes, int n_in,
                              void* d_out, int out_size, void* d_ws, size_t ws_size,
                              hipStream_t stream) {
  (void)n_in; (void)out_size; (void)ws_size;

  const float* theta0 = (const float*)d_in[0];   // [B,N]
  const float* E      = (const float*)d_in[1];   // [B,N,D]
  const float* omega  = (const float*)d_in[2];   // [N]

  const int Nn = in_sizes[2];            // 2048
  const int BN = in_sizes[0];            // B*N
  const int B  = BN / Nn;                // 8
  const int D  = in_sizes[1] / BN;       // 256

  float* theta = (float*)d_out;

  // Workspace layout: C (fp16, B*N*N) | Eh (fp16, B*N*D) | cos (f32) | sin (f32)
  char* ws = (char*)d_ws;
  _Float16* Ch = (_Float16*)ws;
  const size_t Cbytes = (size_t)B * Nn * Nn * sizeof(_Float16);
  _Float16* Eh = (_Float16*)(ws + Cbytes);
  const size_t Ebytes = (size_t)BN * D * sizeof(_Float16);
  float* cf = (float*)(ws + Cbytes + Ebytes);
  float* sf = cf + BN;

  // theta working copy lives in d_out; re-initialized from input every call
  hipMemcpyAsync(theta, theta0, (size_t)BN * sizeof(float),
                 hipMemcpyDeviceToDevice, stream);

  // 1) normalize embeddings -> fp16  (one wave per row, 8 waves per block)
  {
    const int waves  = BN;
    const int blocks = (waves + 7) / 8;
    normalize_rows_f16<<<blocks, 256, 0, stream>>>(E, Eh, BN, D);
  }

  // 2) coupling matrix via WMMA: one wave per 16x64 strip
  {
    const int waves  = B * (Nn >> 4) * (Nn >> 6);
    const int blocks = (waves + 7) / 8;
    coupling_wmma<<<blocks, 256, 0, stream>>>(Eh, Ch, B, Nn, D);
  }

  // 3) 50 sequential Kuramoto steps (two launches each; stream-ordered)
  const float k_over_n = 1.0f / (float)Nn;   // COUPLING_STRENGTH / N
  const float dt = 0.01f;
  for (int s = 0; s < 50; ++s) {
    cossin_kernel<<<(BN + 255) / 256, 256, 0, stream>>>(theta, cf, sf, BN);
    const int waves  = BN;                   // one wave per (b,i) row
    const int blocks = (waves + 7) / 8;
    kuramoto_step<<<blocks, 256, 0, stream>>>(Ch, cf, sf, omega, theta,
                                              B, Nn, k_over_n, dt);
  }
}
